// RecurrentGCN_68650757260096
// MI455X (gfx1250) — matrix operand order repaired and checked by reading, not compiled, not run on hardware
//
#include <hip/hip_runtime.h>
#include <hip/hip_bf16.h>

typedef __attribute__((ext_vector_type(16))) __bf16 v16bf;
typedef __attribute__((ext_vector_type(8)))  float  v8f;

// Branch-free activations: v_exp_f32 is a single TRANS op on CDNA5; the
// libm tanhf expansion costs divergent exec-mask branches per element.
// exp overflow/underflow saturates these identities correctly at +/-inf.
__device__ __forceinline__ float sigmoidf_(float x) { return 1.f / (1.f + __expf(-x)); }
__device__ __forceinline__ float tanhf_(float x)    { return 1.f - 2.f / (__expf(2.f * x) + 1.f); }

// ---------------- zero scratch (mean-aggregation accumulators) ----------------
__global__ __launch_bounds__(256) void k_zero(float* __restrict__ p, int n) {
    int i = blockIdx.x * 256 + threadIdx.x;
    if (i < n) p[i] = 0.f;
}

// ---------------- m = x @ conv_weight  (C=4) ----------------
__global__ __launch_bounds__(256) void k_conv(const float* __restrict__ x,
                                              const float* __restrict__ cw,
                                              float* __restrict__ m, int N) {
    int i = blockIdx.x * 256 + threadIdx.x;
    if (i >= N) return;
    float4 xv = ((const float4*)x)[i];
    float4 o;
    o.x = xv.x * cw[0] + xv.y * cw[4] + xv.z * cw[8]  + xv.w * cw[12];
    o.y = xv.x * cw[1] + xv.y * cw[5] + xv.z * cw[9]  + xv.w * cw[13];
    o.z = xv.x * cw[2] + xv.y * cw[6] + xv.z * cw[10] + xv.w * cw[14];
    o.w = xv.x * cw[3] + xv.y * cw[7] + xv.z * cw[11] + xv.w * cw[15];
    ((float4*)m)[i] = o;
}

// ---------------- edge scatter: summed[dst] += m[src]*w ; cnt[dst] += 1 ----------------
// Streaming-bound: 256 MB of edge data; m (3.2MB) and summed/cnt (4MB) stay L2-resident.
__global__ __launch_bounds__(256) void k_edge(const long long* __restrict__ ei,
                                              const float* __restrict__ ew,
                                              const float* __restrict__ m,
                                              float* __restrict__ summed,
                                              float* __restrict__ cnt, int E) {
    int e = blockIdx.x * 256 + threadIdx.x;
    if (e >= E) return;
    int s = (int)ei[e];
    int d = (int)ei[(size_t)E + e];
    float w = ew[e];
    float4 mv = ((const float4*)m)[s];
    atomicAdd(&summed[d * 4 + 0], mv.x * w);
    atomicAdd(&summed[d * 4 + 1], mv.y * w);
    atomicAdd(&summed[d * 4 + 2], mv.z * w);
    atomicAdd(&summed[d * 4 + 3], mv.w * w);
    atomicAdd(&cnt[d], 1.f);
}

// ---------------- GRUCell(agg, x), C=4, per node ----------------
__global__ __launch_bounds__(256) void k_gru(const float* __restrict__ x,
                                             const float* __restrict__ summed,
                                             const float* __restrict__ cnt,
                                             const float* __restrict__ wih,
                                             const float* __restrict__ whh,
                                             const float* __restrict__ bih,
                                             const float* __restrict__ bhh,
                                             float* __restrict__ hconv, int N) {
    int i = blockIdx.x * 256 + threadIdx.x;
    if (i >= N) return;
    float inv = 1.f / fmaxf(cnt[i], 1.f);
    float4 sv = ((const float4*)summed)[i];
    float agg[4] = { sv.x * inv, sv.y * inv, sv.z * inv, sv.w * inv };
    float4 xv4 = ((const float4*)x)[i];
    float xv[4] = { xv4.x, xv4.y, xv4.z, xv4.w };
    float gi[12], gh[12];
#pragma unroll
    for (int j = 0; j < 12; j++) {
        float a = bih[j], b = bhh[j];
#pragma unroll
        for (int k = 0; k < 4; k++) { a += agg[k] * wih[j * 4 + k]; b += xv[k] * whh[j * 4 + k]; }
        gi[j] = a; gh[j] = b;
    }
    float4 o; float* op = (float*)&o;
#pragma unroll
    for (int c = 0; c < 4; c++) {
        float r  = sigmoidf_(gi[c] + gh[c]);
        float z  = sigmoidf_(gi[4 + c] + gh[4 + c]);
        float nn = tanhf_(gi[8 + c] + r * gh[8 + c]);
        op[c] = (1.f - z) * nn + z * xv[c];
    }
    ((float4*)hconv)[i] = o;
}

// ---------------- Pre-pack LSTM weights into per-lane bf16 WMMA B-fragments ----------------
// Wbig[64][128]: rows 0..3 = W_ih cols, rows 4..35 = W_hh cols, rows 36..63 = 0.
// B k order chosen here (contiguous, same as A): N = lane&15, K = (lane&16) + e (+ 32*chunk).
// bfrag[(tile*2+chunk)*32 + lane][e], 8*2*32*16 bf16 = 16 KB (L2-hot).
__global__ __launch_bounds__(512) void k_bfrag(const float* __restrict__ wih,
                                               const float* __restrict__ whh,
                                               const float* __restrict__ bih,
                                               const float* __restrict__ bhh,
                                               __bf16* __restrict__ bfrag,
                                               float* __restrict__ bsum) {
    int tid = threadIdx.x;        // one block, 512 threads
    int lane = tid & 31;
    int tq = tid >> 5;            // tile*2 + chunk, 0..15
    int q = tq & 1;
    int g = 16 * (tq >> 1) + (lane & 15);   // gate/output column 0..127
#pragma unroll
    for (int e = 0; e < 16; e++) {
        int k = q * 32 + (lane & 16) + e;
        float v = 0.f;
        if (k < 4)       v = wih[g * 4 + k];
        else if (k < 36) v = whh[g * 32 + (k - 4)];
        bfrag[(tq * 32 + lane) * 16 + e] = (__bf16)v;
    }
    if (tid < 128) bsum[tid] = bih[tid] + bhh[tid];
}

// ---------------- LSTM gates via bf16 WMMA + elementwise + linear head ----------------
// One wave = 16 nodes. gates[16,128] = z[16,64(pad)] @ Wbig[64,128]; 16 WMMAs/wave.
__global__ __launch_bounds__(256) void k_lstm(const float* __restrict__ hconv,
                                              const float* __restrict__ h0,
                                              const float* __restrict__ c0,
                                              const __bf16* __restrict__ bfrag,
                                              const float* __restrict__ bsum,
                                              const float* __restrict__ linw,
                                              const float* __restrict__ linb,
                                              float* __restrict__ outp,
                                              float* __restrict__ hout,
                                              float* __restrict__ cout,
                                              int nTiles, int N) {
    __shared__ float zsh[8][1024];   // per-wave 16 nodes x 64 padded K (f32)
    __shared__ float headsh[8][16];  // per-wave per-node linear-head accumulator
    int w = threadIdx.x >> 5, lane = threadIdx.x & 31;
    int gw = blockIdx.x * 8 + w;
    if (gw >= nTiles) return;        // wave-uniform exit: active waves keep EXEC all-ones
    int base = gw * 16;
    float* Z = zsh[w];

    // Stage z = [h_conv(4) | h_prev(32) | zeros] into LDS, 1024 f32 per wave, coalesced.
#pragma unroll
    for (int i = 0; i < 32; i++) {
        int p = i * 32 + lane;
        int mm = p >> 6, k = p & 63;
        int node = base + mm; if (node >= N) node = N - 1;
        float v = 0.f;
        if (k < 4)       v = hconv[node * 4 + k];
        else if (k < 36) v = h0[node * 32 + (k - 4)];
        Z[p] = v;
    }
    if (lane < 16) headsh[w][lane] = linb[0];
    asm volatile("s_wait_dscnt 0" ::: "memory");   // intra-wave LDS ordering (CDNA5 split cnt)

    // A fragments (contiguous k order, same as B): M = lane&15, K = (lane&16) + e
    v16bf a0, a1;
    int mm = lane & 15;
    int kh = lane & 16;
#pragma unroll
    for (int e = 0; e < 16; e++) {
        int k0 = e + kh;
        a0[e] = (__bf16)Z[mm * 64 + k0];
        a1[e] = (__bf16)Z[mm * 64 + 32 + k0];
    }

    const v16bf* Bf = (const v16bf*)bfrag;
    int n = lane & 15;
    int rbase = (lane >> 4) << 3;

#pragma unroll
    for (int t = 0; t < 2; t++) {
        v8f acc[4];                  // i, f, g, o gate tiles for hid = 16t..16t+15
#pragma unroll
        for (int g4 = 0; g4 < 4; g4++) {
            int tt = t + 2 * g4;     // output tile index 0..7  (gate g = 32*g4 + hid)
            float b = bsum[16 * tt + n];
            v8f cacc = { b, b, b, b, b, b, b, b };
            v16bf b0 = Bf[(tt * 2 + 0) * 32 + lane];
            v16bf b1 = Bf[(tt * 2 + 1) * 32 + lane];
            cacc = __builtin_amdgcn_wmma_f32_16x16x32_bf16(false, a0, false, b0,
                                                           (short)0, cacc, false, false);
            cacc = __builtin_amdgcn_wmma_f32_16x16x32_bf16(false, a1, false, b1,
                                                           (short)0, cacc, false, false);
            acc[g4] = cacc;
        }
        int hid = 16 * t + n;
        float lw = linw[hid];
        // D layout: VGPR r -> node = base + (lane>=16 ? 8 : 0) + r, column = hid
#pragma unroll
        for (int r = 0; r < 8; r++) {
            int node = base + rbase + r;
            float ig = sigmoidf_(acc[0][r]);
            float fg = sigmoidf_(acc[1][r]);
            float gg = tanhf_(acc[2][r]);
            float og = sigmoidf_(acc[3][r]);
            if (node < N) {
                float cp = c0[node * 32 + hid];
                float cn = fg * cp + ig * gg;
                float hn = og * tanhf_(cn);
                hout[node * 32 + hid] = hn;
                cout[node * 32 + hid] = cn;
                float rl = hn > 0.f ? hn : 0.f;
                atomicAdd(&headsh[w][rbase + r], rl * lw);   // ds_add_f32
            }
        }
    }
    asm volatile("s_wait_dscnt 0" ::: "memory");
    if (lane < 16 && base + lane < N) outp[base + lane] = headsh[w][lane];
}

extern "C" void kernel_launch(void* const* d_in, const int* in_sizes, int n_in,
                              void* d_out, int out_size, void* d_ws, size_t ws_size,
                              hipStream_t stream) {
    const float*     x    = (const float*)d_in[0];
    const long long* ei   = (const long long*)d_in[1];
    const float*     ew   = (const float*)d_in[2];
    const float*     h0   = (const float*)d_in[3];
    const float*     c0   = (const float*)d_in[4];
    const float*     cw   = (const float*)d_in[5];
    const float*     gwih = (const float*)d_in[6];
    const float*     gwhh = (const float*)d_in[7];
    const float*     gbih = (const float*)d_in[8];
    const float*     gbhh = (const float*)d_in[9];
    const float*     lwih = (const float*)d_in[10];
    const float*     lwhh = (const float*)d_in[11];
    const float*     lbih = (const float*)d_in[12];
    const float*     lbhh = (const float*)d_in[13];
    const float*     linw = (const float*)d_in[14];
    const float*     linb = (const float*)d_in[15];

    int N = in_sizes[0] / 4;
    int E = in_sizes[2];

    // Workspace layout (floats): m[4N] | summed[4N] | cnt[N] | hconv[4N] | bfrag(bf16,8192) | bsum[128]
    float* d_m    = (float*)d_ws;
    float* d_sum  = d_m + (size_t)N * 4;
    float* d_cnt  = d_sum + (size_t)N * 4;
    float* d_hc   = d_cnt + (size_t)N;
    size_t fOff   = (size_t)N * 13;
    fOff = (fOff + 15) & ~(size_t)15;              // 64B alignment for b128 fragment loads
    __bf16* d_bf  = (__bf16*)(d_m + fOff);
    float* d_bsum = (float*)(d_bf + 8192);

    float* outp  = (float*)d_out;
    float* houtp = outp + (size_t)N;
    float* coutp = houtp + (size_t)N * 32;

    int nz = N * 5;                                 // summed + cnt
    hipLaunchKernelGGL(k_zero,  dim3((nz + 255) / 256), dim3(256), 0, stream, d_sum, nz);
    hipLaunchKernelGGL(k_conv,  dim3((N + 255) / 256),  dim3(256), 0, stream, x, cw, d_m, N);
    hipLaunchKernelGGL(k_bfrag, dim3(1),                dim3(512), 0, stream,
                       lwih, lwhh, lbih, lbhh, d_bf, d_bsum);
    hipLaunchKernelGGL(k_edge,  dim3((E + 255) / 256),  dim3(256), 0, stream,
                       ei, ew, d_m, d_sum, d_cnt, E);
    hipLaunchKernelGGL(k_gru,   dim3((N + 255) / 256),  dim3(256), 0, stream,
                       x, d_sum, d_cnt, gwih, gwhh, gbih, gbhh, d_hc, N);
    int nTiles = (N + 15) / 16;
    hipLaunchKernelGGL(k_lstm,  dim3((nTiles + 7) / 8), dim3(256), 0, stream,
                       d_hc, h0, c0, d_bf, d_bsum, linw, linb,
                       outp, houtp, coutp, nTiles, N);
}
